// FusedLinearListNetLoss_884763263402
// MI455X (gfx1250) — matrix operand order repaired and checked by reading, not compiled, not run on hardware
//
#include <hip/hip_runtime.h>
#include <hip/hip_bf16.h>
#include <math.h>

typedef __attribute__((ext_vector_type(16))) __bf16 v16bf;
typedef __attribute__((ext_vector_type(8)))  float  v8f;
typedef __attribute__((ext_vector_type(4)))  float  f4v;

#define NROWS 4096
#define HDIM  512
#define VOCAB 50257
#define MT 32    // rows per block
#define NT 128   // vocab cols per chunk (2 x 16 cols per wave)
#define KT 32    // K per WMMA step

// LDS byte offset of a __shared__ object (generic -> addrspace(3) -> int)
#define LDS_OFF(p) ((unsigned)(unsigned long long)(__attribute__((address_space(3))) const void*)(const void*)(p))

__device__ __forceinline__ void async_load_b64(unsigned lds, unsigned voff,
                                               unsigned long long base) {
  asm volatile("global_load_async_to_lds_b64 %0, %1, %2"
               :: "v"(lds), "v"(voff), "s"(base) : "memory");
}
__device__ __forceinline__ void async_load_b128(unsigned lds, unsigned voff,
                                                unsigned long long base) {
  asm volatile("global_load_async_to_lds_b128 %0, %1, %2"
               :: "v"(lds), "v"(voff), "s"(base) : "memory");
}
__device__ __forceinline__ void wait_async0() {
  asm volatile("s_wait_asynccnt 0x0" ::: "memory");
}

// ---------------------------------------------------------------------------
// Kernel 0: x -> bf16 hi/lo split; also zero the loss accumulator
// ---------------------------------------------------------------------------
__global__ void k_xconv(const float* __restrict__ x, __bf16* __restrict__ xh,
                        __bf16* __restrict__ xl, float* __restrict__ acc) {
  size_t i = (size_t)blockIdx.x * 256 + threadIdx.x;
  if (i == 0) *acc = 0.0f;
  if (i < (size_t)NROWS * HDIM) {
    float v = x[i];
    __bf16 h = (__bf16)v;
    xh[i] = h;
    xl[i] = (__bf16)(v - (float)h);
  }
}

// ---------------------------------------------------------------------------
// Kernel 0b (big-ws path): weight -> bf16 hi/lo split (L2-resident operand)
// ---------------------------------------------------------------------------
__global__ void k_wconv(const float* __restrict__ wsrc, __bf16* __restrict__ wh,
                        __bf16* __restrict__ wl) {
  size_t i = (size_t)blockIdx.x * 256 + threadIdx.x;
  if (i < (size_t)VOCAB * HDIM) {
    float v = wsrc[i];
    __bf16 h = (__bf16)v;
    wh[i] = h;
    wl[i] = (__bf16)(v - (float)h);
  }
}

// ---------------------------------------------------------------------------
// Kernel 1: per-row online logsumexp of targets (NaN row -> +inf), NT loads
// ---------------------------------------------------------------------------
__global__ __launch_bounds__(256) void k_tstats(const float* __restrict__ targets,
                                                float* __restrict__ lse_t) {
  const int n = blockIdx.x;
  const int t = threadIdx.x;
  const float* row = targets + (size_t)n * VOCAB;
  float m = -INFINITY, s = 0.0f;
  int nanf = 0;
  const int nv4 = VOCAB >> 2;
  for (int i = t; i < nv4; i += 256) {
    f4v v = __builtin_nontemporal_load((const f4v*)row + i);
#pragma unroll
    for (int j = 0; j < 4; ++j) {
      float xv = v[j];
      nanf |= (xv != xv);
      if (xv > m) { s *= __expf(m - xv); m = xv; s += 1.0f; }
      else        { s += __expf(xv - m); }
    }
  }
  if (t == 0) {
    for (int i = nv4 * 4; i < VOCAB; ++i) {
      float xv = row[i];
      nanf |= (xv != xv);
      if (xv > m) { s *= __expf(m - xv); m = xv; s += 1.0f; }
      else        { s += __expf(xv - m); }
    }
  }
  __shared__ float rm[256], rs[256];
  __shared__ int   rn[256];
  rm[t] = m; rs[t] = s; rn[t] = nanf;
  __syncthreads();
  for (int off = 128; off > 0; off >>= 1) {
    if (t < off) {
      rn[t] |= rn[t + off];
      float m2 = rm[t + off], s2 = rs[t + off];
      float m1 = rm[t],       s1 = rs[t];
      if (s2 > 0.0f) {
        if (m2 > m1) { s1 *= __expf(m1 - m2); m1 = m2; }
        s1 += s2 * __expf(m2 - m1);
        rm[t] = m1; rs[t] = s1;
      }
    }
    __syncthreads();
  }
  if (t == 0) lse_t[n] = rn[0] ? INFINITY : (rm[0] + __logf(rs[0]));
}

// ---------------------------------------------------------------------------
// Kernel 2: fused split-bf16 WMMA GEMM + streamed ListNet loss reduction
//   WSPLIT=true : A and B staged with GLOBAL_LOAD_ASYNC_TO_LDS (pure copies)
//   WSPLIT=false: B converted fp32->bf16 hi/lo on the fly (small-ws fallback)
// ---------------------------------------------------------------------------
template <bool WSPLIT>
__global__ __launch_bounds__(256) void k_main(const float* __restrict__ weight,
                                              const float* __restrict__ bias,
                                              const float* __restrict__ targets,
                                              const __bf16* __restrict__ xh,
                                              const __bf16* __restrict__ xl,
                                              const __bf16* __restrict__ wh,
                                              const __bf16* __restrict__ wl,
                                              const float* __restrict__ lse_t,
                                              float* __restrict__ acc) {
  __shared__ __attribute__((aligned(32))) __bf16 sA[2][2][MT * KT];  // 8 KB
  __shared__ __attribute__((aligned(32))) __bf16 sB[2][2][NT * KT];  // 32 KB
  __shared__ float sred[8][2][8][4];                                 // 2 KB

  const int t    = threadIdx.x;
  const int lane = t & 31;
  const int w    = t >> 5;
  const int mw   = w >> 2;   // 0..1 (M tile)
  const int nw   = w & 3;    // 0..3 (pair of N tiles)
  const int row0 = blockIdx.x * MT;
  const int lh   = lane >> 4;
  const int ln15 = lane & 15;
  const int Mbase = mw * 16 + lh * 8;

  float lse_r[8];
  const float* tRow[8];
#pragma unroll
  for (int r = 0; r < 8; ++r) {
    int gr = row0 + Mbase + r;
    lse_r[r] = lse_t[gr];
    tRow[r]  = targets + (size_t)gr * VOCAB;
  }
  float mrun[8], srun[8], drun[8], sprun[8];
#pragma unroll
  for (int r = 0; r < 8; ++r) { mrun[r] = -INFINITY; srun[r] = 0.f; drun[r] = 0.f; sprun[r] = 0.f; }

  // WMMA fragment addressing (A 16x32 / B 32x16, contiguous k = 16*(lane/16) + s)
  const int Am  = ln15 + mw * 16;
  const int Akb = lh << 4;   // 0 or 16
  const int Bkb = lh << 4;   // 0 or 16

  // staging indices
  const int ar = t >> 3, akk = (t & 7) << 2;   // A: 4 elements (8B) per matrix
  const int bc = t >> 1, bkk = (t & 1) << 4;   // B: 16 elements (32B) per matrix

  const unsigned long long xhB = (unsigned long long)xh;
  const unsigned long long xlB = (unsigned long long)xl;
  const unsigned long long whB = (unsigned long long)wh;
  const unsigned long long wlB = (unsigned long long)wl;

  const int nchunk = (VOCAB + NT - 1) / NT;  // 393
  for (int ch = 0; ch < nchunk; ++ch) {
    const int v0 = ch * NT;

    auto stage = [&](int buf, int ks) {
      const int k0 = ks * KT;
      {  // A hi/lo: pure async copy from ws
        unsigned go = (unsigned)(((row0 + ar) * HDIM + k0 + akk) * 2);
        async_load_b64(LDS_OFF(&sA[buf][0][ar * KT + akk]), go, xhB);
        async_load_b64(LDS_OFF(&sA[buf][1][ar * KT + akk]), go, xlB);
      }
      int vc = v0 + bc;
      if constexpr (WSPLIT) {  // B hi/lo: pure async copy from pre-split weight
        if (vc >= VOCAB) vc = 0;  // clamp; masked in epilogue
        unsigned go = (unsigned)((vc * HDIM + k0 + bkk) * 2);
        unsigned dh = LDS_OFF(&sB[buf][0][bc * KT + bkk]);
        unsigned dl = LDS_OFF(&sB[buf][1][bc * KT + bkk]);
        async_load_b128(dh,      go,      whB);
        async_load_b128(dh + 16, go + 16, whB);
        async_load_b128(dl,      go,      wlB);
        async_load_b128(dl + 16, go + 16, wlB);
      } else {  // B: fp32 -> bf16 hi/lo split through VGPRs
        float wv[16];
        if (vc < VOCAB) {
          const float* wp = &weight[(size_t)vc * HDIM + k0 + bkk];
          f4v q0 = *(const f4v*)(wp + 0);
          f4v q1 = *(const f4v*)(wp + 4);
          f4v q2 = *(const f4v*)(wp + 8);
          f4v q3 = *(const f4v*)(wp + 12);
#pragma unroll
          for (int j = 0; j < 4; ++j) {
            wv[j] = q0[j]; wv[4 + j] = q1[j]; wv[8 + j] = q2[j]; wv[12 + j] = q3[j];
          }
        } else {
#pragma unroll
          for (int j = 0; j < 16; ++j) wv[j] = 0.0f;
        }
        __bf16* dh = &sB[buf][0][bc * KT + bkk];
        __bf16* dl = &sB[buf][1][bc * KT + bkk];
#pragma unroll
        for (int j = 0; j < 16; ++j) {
          __bf16 hb = (__bf16)wv[j];
          dh[j] = hb;
          dl[j] = (__bf16)(wv[j] - (float)hb);
        }
      }
    };

    v8f zero = {};
    v8f accv[2];
    accv[0] = zero; accv[1] = zero;

    stage(0, 0);
    wait_async0();
    __syncthreads();

    for (int ks = 0; ks < HDIM / KT; ++ks) {
      const int cur = ks & 1;
      if (ks + 1 < HDIM / KT) stage(cur ^ 1, ks + 1);  // overlap with compute

      v16bf ah, al;
      const __bf16* pah = &sA[cur][0][Am * KT + Akb];
      const __bf16* pal = &sA[cur][1][Am * KT + Akb];
#pragma unroll
      for (int i = 0; i < 8; ++i) {
        ah[i] = pah[i];          al[i] = pal[i];
        ah[8 + i] = pah[8 + i]; al[8 + i] = pal[8 + i];
      }
#pragma unroll
      for (int q = 0; q < 2; ++q) {
        const int cb = (nw * 32 + q * 16 + ln15) * KT + Bkb;
        v16bf bh = *(const v16bf*)&sB[cur][0][cb];
        v16bf bl = *(const v16bf*)&sB[cur][1][cb];
        // split-bf16: hi*hi + lo*hi + hi*lo (~fp32 logits), fp32 accumulate
        accv[q] = __builtin_amdgcn_wmma_f32_16x16x32_bf16(false, ah, false, bh, (short)0, accv[q], false, false);
        accv[q] = __builtin_amdgcn_wmma_f32_16x16x32_bf16(false, al, false, bh, (short)0, accv[q], false, false);
        accv[q] = __builtin_amdgcn_wmma_f32_16x16x32_bf16(false, ah, false, bl, (short)0, accv[q], false, false);
      }
      if (ks + 1 < HDIM / KT) { wait_async0(); __syncthreads(); }
    }

    // streamed epilogue: online lse(logits), dot(p_target, logits), sum(p)
#pragma unroll
    for (int q = 0; q < 2; ++q) {
      const int col = v0 + nw * 32 + q * 16 + ln15;
      if (col < VOCAB) {
        const float bcol = bias[col];
#pragma unroll
        for (int r = 0; r < 8; ++r) {
          float logit = accv[q][r] + bcol;
          if (logit > mrun[r]) { srun[r] *= __expf(mrun[r] - logit); mrun[r] = logit; srun[r] += 1.0f; }
          else                 { srun[r] += __expf(logit - mrun[r]); }
          float tv = __builtin_nontemporal_load(tRow[r] + col);
          float p  = (tv != tv) ? 0.0f : __expf(tv - lse_r[r]);
          drun[r]  += p * logit;
          sprun[r] += p;
        }
      }
    }
  }

  // reduce the 16 column-lanes of each half-wave with shuffles (wave32)
#pragma unroll
  for (int r = 0; r < 8; ++r) {
    float m = mrun[r], s = srun[r], d = drun[r], sp = sprun[r];
#pragma unroll
    for (int mask = 1; mask <= 8; mask <<= 1) {
      float m2  = __shfl_xor(m, mask, 32);
      float s2  = __shfl_xor(s, mask, 32);
      float d2  = __shfl_xor(d, mask, 32);
      float sp2 = __shfl_xor(sp, mask, 32);
      d += d2; sp += sp2;
      if (s2 > 0.0f) {
        if (m2 > m) { s *= __expf(m - m2); m = m2; }
        s += s2 * __expf(m2 - m);
      }
    }
    if (ln15 == 0) { sred[w][lh][r][0] = m; sred[w][lh][r][1] = s; sred[w][lh][r][2] = d; sred[w][lh][r][3] = sp; }
  }
  __syncthreads();
  if (t < MT) {  // one thread per row: combine the 4 N-group partials
    const int rmw = t >> 4, mi = t & 15, rr = mi & 7, rlh = mi >> 3;
    float m = -INFINITY, s = 0.0f, d = 0.0f, sp = 0.0f;
    for (int rnw = 0; rnw < 4; ++rnw) {
      const float* p = sred[rmw * 4 + rnw][rlh][rr];
      float m2 = p[0], s2 = p[1];
      d += p[2]; sp += p[3];
      if (s2 > 0.0f) {
        if (m2 > m) { s *= __expf(m - m2); m = m2; }
        s += s2 * __expf(m2 - m);
      }
    }
    float lse = m + __logf(s);
    atomicAdd(acc, lse * sp - d);  // loss_row = lse_logits * sum(p) - dot(p, logits)
  }
}

// ---------------------------------------------------------------------------
// Kernel 3: finalize mean
// ---------------------------------------------------------------------------
__global__ void k_fin(const float* __restrict__ acc, float* __restrict__ out) {
  if (threadIdx.x == 0 && blockIdx.x == 0) out[0] = acc[0] * (1.0f / (float)NROWS);
}

extern "C" void kernel_launch(void* const* d_in, const int* in_sizes, int n_in,
                              void* d_out, int out_size, void* d_ws, size_t ws_size,
                              hipStream_t stream) {
  (void)in_sizes; (void)n_in; (void)out_size;
  const float* x       = (const float*)d_in[0];
  const float* targets = (const float*)d_in[1];
  const float* weight  = (const float*)d_in[2];
  const float* bias    = (const float*)d_in[3];
  float* out = (float*)d_out;

  // ws layout: [acc:16B][lse_t: N f32][x_hi][x_lo][w_hi][w_lo (big path only)]
  char* ws = (char*)d_ws;
  float*  acc   = (float*)ws;
  float*  lse_t = (float*)(ws + 16);
  __bf16* xh    = (__bf16*)(ws + 16 + (size_t)NROWS * 4);
  __bf16* xl    = xh + (size_t)NROWS * HDIM;
  __bf16* wh    = xl + (size_t)NROWS * HDIM;
  __bf16* wl    = wh + (size_t)VOCAB * HDIM;
  const size_t need_big = 16 + (size_t)NROWS * 4 + 2ull * NROWS * HDIM * 2 +
                          2ull * VOCAB * HDIM * 2;  // ~111 MB
  const bool big = ws_size >= need_big;

  k_xconv<<<(NROWS * HDIM + 255) / 256, 256, 0, stream>>>(x, xh, xl, acc);
  k_tstats<<<NROWS, 256, 0, stream>>>(targets, lse_t);
  if (big) {
    k_wconv<<<(int)(((size_t)VOCAB * HDIM + 255) / 256), 256, 0, stream>>>(weight, wh, wl);
    k_main<true><<<NROWS / MT, 256, 0, stream>>>(weight, bias, targets, xh, xl, wh, wl, lse_t, acc);
  } else {
    k_main<false><<<NROWS / MT, 256, 0, stream>>>(weight, bias, targets, xh, xl, nullptr, nullptr, lse_t, acc);
  }
  k_fin<<<1, 64, 0, stream>>>(acc, out);
}
